// AttentionBlock_8392366096901
// MI455X (gfx1250) — compile-verified, matrix operand order checked by reading, not run on hardware
//
#include <hip/hip_runtime.h>
#include <hip/hip_bf16.h>
#include <stdint.h>

// ---------------------------------------------------------------------------
// AttentionBlock for MI455X (gfx1250): GroupNorm -> QKV -> MHSA -> proj + res
// All GEMMs use v_wmma_f32_16x16x32_f16 (wave32 WMMA, f32 accumulate).
// GEMM kernels use 32x64 per-wave tiles (8 WMMAs / 6 fragment loads per
// k-step) inside a 128x128 block tile for L0/L2 reuse.
// ---------------------------------------------------------------------------

typedef __attribute__((ext_vector_type(16))) _Float16 v16h;
typedef __attribute__((ext_vector_type(8)))  float    v8f;

#define T_DIM 2304      // 48*48
#define C_DIM 512
#define NHEAD 8
#define DHEAD 64
#define QK_SCALE 0.35355339059327373f   // 64^(-1/4)

union FragU { uint32_t u[8]; v16h v; };

// Load a 16x32 f16 WMMA A/B fragment. `p` points to a matrix where the outer
// index (M for A, N for B) has row stride `ld` halfs and K is contiguous.
// CDNA5 16-bit layout: lanes 0-15 hold outer=lane, K halves {0..7,16..23};
// lanes 16-31 hold outer=lane-16, K halves {8..15,24..31}; each VGPR = K pair.
__device__ __forceinline__ v16h load_frag16x32(const _Float16* p, int outer0,
                                               int k0, int ld) {
  int lane = threadIdx.x & 31;
  int oi   = lane & 15;
  int d0   = (lane >> 4) << 2;             // dword offset 0 or 4
  const uint32_t* p32 =
      reinterpret_cast<const uint32_t*>(p + (size_t)(outer0 + oi) * ld + k0);
  FragU r;
#pragma unroll
  for (int i = 0; i < 4; ++i) {
    r.u[i]     = p32[d0 + i];              // K = 2*(d0+i)   .. +1
    r.u[i + 4] = p32[8 + d0 + i];          // K = 16+2*(d0+i)..+1
  }
  return r.v;
}

__device__ __forceinline__ v8f wmma_f16(v16h a, v16h b, v8f c) {
  return __builtin_amdgcn_wmma_f32_16x16x32_f16(false, a, false, b,
                                                (short)0, c, false, false);
}

// Row-wise reductions across the 16 lanes that share a D-matrix row group.
__device__ __forceinline__ float rowmax16(float v) {
#pragma unroll
  for (int off = 1; off < 16; off <<= 1) v = fmaxf(v, __shfl_xor(v, off, 32));
  return v;
}
__device__ __forceinline__ float rowsum16(float v) {
#pragma unroll
  for (int off = 1; off < 16; off <<= 1) v += __shfl_xor(v, off, 32);
  return v;
}

// ---------------------------------------------------------------------------
// Kernel 1: convert qkv_w (1536x512) and proj_w (512x512) f32 -> f16
// ---------------------------------------------------------------------------
__global__ void cvt_w_kernel(const float* __restrict__ qw,
                             const float* __restrict__ pw,
                             _Float16* __restrict__ wq,
                             _Float16* __restrict__ wp) {
  int i = blockIdx.x * 256 + threadIdx.x;
  if (i < 3 * C_DIM * C_DIM) wq[i] = (_Float16)qw[i];
  if (i < C_DIM * C_DIM)     wp[i] = (_Float16)pw[i];
}

// ---------------------------------------------------------------------------
// Kernel 2: GroupNorm. One block per (b, group of 16 channels).
// Writes hnT[b][t][c] f16 (transposed so GEMM K-dim is contiguous).
// ---------------------------------------------------------------------------
__global__ __launch_bounds__(256) void gn_kernel(const float* __restrict__ x,
                                                 const float* __restrict__ sc,
                                                 const float* __restrict__ bi,
                                                 _Float16* __restrict__ hnT) {
  const int GC = 16;                       // channels per group (512/32)
  int b = blockIdx.x >> 5;
  int g = blockIdx.x & 31;
  const float* xp = x + ((size_t)b * C_DIM + (size_t)g * GC) * T_DIM;

  float s = 0.f, ss = 0.f;
  for (int i = threadIdx.x; i < GC * T_DIM; i += 256) {
    float v = xp[i];
    s += v; ss += v * v;
  }
  __shared__ float r1[256], r2[256];
  r1[threadIdx.x] = s; r2[threadIdx.x] = ss;
  __syncthreads();
  for (int off = 128; off > 0; off >>= 1) {
    if (threadIdx.x < off) {
      r1[threadIdx.x] += r1[threadIdx.x + off];
      r2[threadIdx.x] += r2[threadIdx.x + off];
    }
    __syncthreads();
  }
  const float inv_n = 1.0f / (float)(GC * T_DIM);
  float mean = r1[0] * inv_n;
  float var  = r2[0] * inv_n - mean * mean;
  float rstd = rsqrtf(var + 1e-5f);

  // Normalize: channel-minor iteration so hnT writes are contiguous f16.
  for (int i = threadIdx.x; i < GC * T_DIM; i += 256) {
    int t  = i >> 4;
    int cl = i & 15;
    int cg = g * GC + cl;
    float v = (xp[(size_t)cl * T_DIM + t] - mean) * rstd * sc[cg] + bi[cg];
    hnT[((size_t)b * T_DIM + t) * C_DIM + cg] = (_Float16)v;
  }
}

// ---------------------------------------------------------------------------
// Kernel 3: QKV GEMM.  out[b,o,t] = sum_c wq[o,c]*hnT[b,t,c] + qkv_b[o]
// Scatter: q,k -> [bh][t][d] f16 (scaled by d^-1/4), v -> [bh][d][t] f16.
// Block tile 128(M)x128(N); wave tile 32x64 (2 A-frags x 4 B-frags -> 8 WMMA).
// grid = (T/128, 1536/128, B) = (18, 12, 4).
// ---------------------------------------------------------------------------
__global__ __launch_bounds__(256) void qkv_gemm_kernel(
    const _Float16* __restrict__ wq, const _Float16* __restrict__ hnT,
    const float* __restrict__ qkvb, _Float16* __restrict__ qT,
    _Float16* __restrict__ kT, _Float16* __restrict__ vv) {
  int wave = threadIdx.x >> 5, lane = threadIdx.x & 31;
  int b  = blockIdx.z;
  int m0 = blockIdx.y * 128 + (wave & 3) * 32;         // o
  int n0 = blockIdx.x * 128 + (wave >> 2) * 64;        // t
  const _Float16* Bp = hnT + (size_t)b * T_DIM * C_DIM;

  v8f acc[2][4] = {};
  for (int k0 = 0; k0 < C_DIM; k0 += 32) {
    __builtin_prefetch(wq + (size_t)m0 * C_DIM + k0 + 64, 0, 3);
    __builtin_prefetch(Bp + (size_t)n0 * C_DIM + k0 + 64, 0, 3);
    v16h a0 = load_frag16x32(wq, m0, k0, C_DIM);
    v16h a1 = load_frag16x32(wq, m0 + 16, k0, C_DIM);
#pragma unroll
    for (int j = 0; j < 4; ++j) {
      v16h bf = load_frag16x32(Bp, n0 + 16 * j, k0, C_DIM);
      acc[0][j] = wmma_f16(a0, bf, acc[0][j]);
      acc[1][j] = wmma_f16(a1, bf, acc[1][j]);
    }
  }

  int hi = (lane >> 4) << 3;
  int nn = lane & 15;
#pragma unroll
  for (int i = 0; i < 2; ++i) {
#pragma unroll
    for (int j = 0; j < 4; ++j) {
      int n = n0 + 16 * j + nn;                        // t
#pragma unroll
      for (int r = 0; r < 8; ++r) {
        int m = m0 + 16 * i + r + hi;                  // o in [0,1536)
        float val = acc[i][j][r] + qkvb[m];
        int s3  = m >> 9;                              // 0=q 1=k 2=v
        int rem = m & 511;
        int h   = rem >> 6;
        int c   = rem & 63;
        size_t bh = (size_t)(b * NHEAD + h);
        if (s3 == 0)
          qT[(bh * T_DIM + n) * DHEAD + c] = (_Float16)(val * QK_SCALE);
        else if (s3 == 1)
          kT[(bh * T_DIM + n) * DHEAD + c] = (_Float16)(val * QK_SCALE);
        else
          vv[(bh * DHEAD + c) * T_DIM + n] = (_Float16)val;
      }
    }
  }
}

// ---------------------------------------------------------------------------
// Kernel 4: flash attention.  One wave owns a 16-row query tile; loops over
// all key/value columns in chunks of 32 with online softmax.  P is staged
// through a per-wave LDS tile to convert D-layout -> A-layout.
// grid = 32 bh * 18 = 576 blocks of 256 threads.
// ---------------------------------------------------------------------------
__global__ __launch_bounds__(256) void attn_kernel(
    const _Float16* __restrict__ qT, const _Float16* __restrict__ kT,
    const _Float16* __restrict__ vv, _Float16* __restrict__ aT) {
  const int LDP = 34;                                  // LDS row stride (halfs)
  __shared__ _Float16 plds[8][16 * LDP];

  int wave = threadIdx.x >> 5, lane = threadIdx.x & 31;
  int bh   = blockIdx.x / 18;
  int blk  = blockIdx.x % 18;
  int qt0  = (blk * 8 + wave) * 16;                    // query row base
  int b = bh >> 3, h = bh & 7;

  const _Float16* qb = qT + (size_t)bh * T_DIM * DHEAD;
  const _Float16* kb = kT + (size_t)bh * T_DIM * DHEAD;
  const _Float16* vb = vv + (size_t)bh * DHEAD * T_DIM;
  _Float16* pl = plds[wave];

  // Query fragments (kept resident for the whole s loop)
  v16h aq0 = load_frag16x32(qb, qt0, 0, DHEAD);
  v16h aq1 = load_frag16x32(qb, qt0, 32, DHEAD);

  v8f o[4] = {v8f{}, v8f{}, v8f{}, v8f{}};
  float m_r[8], l_r[8];
#pragma unroll
  for (int r = 0; r < 8; ++r) { m_r[r] = -1e30f; l_r[r] = 0.f; }

  int hi = (lane >> 4) << 3;
  int nn = lane & 15;

  for (int s0 = 0; s0 < T_DIM; s0 += 32) {
    // S = Q^T K for two 16-col subtiles (D rows = t, cols = s)
    v8f sA = {}, sB = {};
    {
      v16h bk0 = load_frag16x32(kb, s0, 0, DHEAD);
      v16h bk1 = load_frag16x32(kb, s0, 32, DHEAD);
      sA = wmma_f16(aq0, bk0, sA);
      sA = wmma_f16(aq1, bk1, sA);
      bk0 = load_frag16x32(kb, s0 + 16, 0, DHEAD);
      bk1 = load_frag16x32(kb, s0 + 16, 32, DHEAD);
      sB = wmma_f16(aq0, bk0, sB);
      sB = wmma_f16(aq1, bk1, sB);
    }
    // Online softmax per row; P -> LDS (f16, 16x32 tile)
#pragma unroll
    for (int r = 0; r < 8; ++r) {
      float a = sA[r], bv = sB[r];
      float mx   = rowmax16(fmaxf(a, bv));
      float mnew = fmaxf(m_r[r], mx);
      float corr = __expf(m_r[r] - mnew);
      float pa   = __expf(a - mnew);
      float pb   = __expf(bv - mnew);
      l_r[r] = l_r[r] * corr + rowsum16(pa + pb);
      m_r[r] = mnew;
#pragma unroll
      for (int ct = 0; ct < 4; ++ct) o[ct][r] *= corr;
      int row = r + hi;
      pl[row * LDP + nn]      = (_Float16)pa;
      pl[row * LDP + nn + 16] = (_Float16)pb;
    }
    // O += P * V^T  (A from LDS, B from v[c][t] with K=s contiguous)
    v16h pf = load_frag16x32(pl, 0, 0, LDP);
#pragma unroll
    for (int ct = 0; ct < 4; ++ct) {
      v16h bvf = load_frag16x32(vb, ct * 16, s0, T_DIM);
      o[ct] = wmma_f16(pf, bvf, o[ct]);
    }
  }

  // Normalize and write aT[b][t][cglobal] f16 (K-contiguous for proj GEMM)
#pragma unroll
  for (int r = 0; r < 8; ++r) {
    float linv = 1.0f / l_r[r];
    int row = qt0 + r + hi;
#pragma unroll
    for (int ct = 0; ct < 4; ++ct) {
      int cg = h * DHEAD + ct * 16 + nn;
      aT[((size_t)b * T_DIM + row) * C_DIM + cg] = (_Float16)(o[ct][r] * linv);
    }
  }
}

// ---------------------------------------------------------------------------
// Kernel 5: proj GEMM + bias + residual.
// out[b,o,t] = x[b,o,t] + proj_b[o] + sum_c wp[o,c]*aT[b,t,c]
// Block tile 128x128; wave tile 32x64.  grid = (18, 4, 4).
// ---------------------------------------------------------------------------
__global__ __launch_bounds__(256) void proj_gemm_kernel(
    const _Float16* __restrict__ wp, const _Float16* __restrict__ aT,
    const float* __restrict__ pb, const float* __restrict__ x,
    float* __restrict__ out) {
  int wave = threadIdx.x >> 5, lane = threadIdx.x & 31;
  int b  = blockIdx.z;
  int m0 = blockIdx.y * 128 + (wave & 3) * 32;         // o
  int n0 = blockIdx.x * 128 + (wave >> 2) * 64;        // t
  const _Float16* Bp = aT + (size_t)b * T_DIM * C_DIM;

  v8f acc[2][4] = {};
  for (int k0 = 0; k0 < C_DIM; k0 += 32) {
    __builtin_prefetch(wp + (size_t)m0 * C_DIM + k0 + 64, 0, 3);
    __builtin_prefetch(Bp + (size_t)n0 * C_DIM + k0 + 64, 0, 3);
    v16h a0 = load_frag16x32(wp, m0, k0, C_DIM);
    v16h a1 = load_frag16x32(wp, m0 + 16, k0, C_DIM);
#pragma unroll
    for (int j = 0; j < 4; ++j) {
      v16h bf = load_frag16x32(Bp, n0 + 16 * j, k0, C_DIM);
      acc[0][j] = wmma_f16(a0, bf, acc[0][j]);
      acc[1][j] = wmma_f16(a1, bf, acc[1][j]);
    }
  }

  int hi = (lane >> 4) << 3;
  int nn = lane & 15;
#pragma unroll
  for (int i = 0; i < 2; ++i) {
#pragma unroll
    for (int j = 0; j < 4; ++j) {
      int n = n0 + 16 * j + nn;
#pragma unroll
      for (int r = 0; r < 8; ++r) {
        int m = m0 + 16 * i + r + hi;
        size_t idx = ((size_t)b * C_DIM + m) * T_DIM + n;
        out[idx] = x[idx] + pb[m] + acc[i][j][r];
      }
    }
  }
}

// ---------------------------------------------------------------------------
extern "C" void kernel_launch(void* const* d_in, const int* in_sizes, int n_in,
                              void* d_out, int out_size, void* d_ws,
                              size_t ws_size, hipStream_t stream) {
  const float* x     = (const float*)d_in[0];
  const float* gns   = (const float*)d_in[1];
  const float* gnb   = (const float*)d_in[2];
  const float* qkvw  = (const float*)d_in[3];
  const float* qkvb  = (const float*)d_in[4];
  const float* projw = (const float*)d_in[5];
  const float* projb = (const float*)d_in[6];
  (void)d_in[7];  // num_heads == 8 (compile-time)
  float* out = (float*)d_out;

  // Workspace layout (256B aligned slices)
  char* ws = (char*)d_ws;
  size_t off = 0;
  auto take = [&](size_t bytes) {
    char* p = ws + off;
    off += (bytes + 255) & ~(size_t)255;
    return p;
  };
  _Float16* wq  = (_Float16*)take((size_t)3 * C_DIM * C_DIM * 2);       // 1.5 MB
  _Float16* wp  = (_Float16*)take((size_t)C_DIM * C_DIM * 2);           // 0.5 MB
  _Float16* hnT = (_Float16*)take((size_t)4 * T_DIM * C_DIM * 2);       // 9.4 MB
  _Float16* qT  = (_Float16*)take((size_t)4 * NHEAD * T_DIM * DHEAD * 2);
  _Float16* kT  = (_Float16*)take((size_t)4 * NHEAD * T_DIM * DHEAD * 2);
  _Float16* vv  = (_Float16*)take((size_t)4 * NHEAD * DHEAD * T_DIM * 2);
  _Float16* aT  = (_Float16*)take((size_t)4 * T_DIM * C_DIM * 2);
  (void)ws_size;

  // 1. weights -> f16
  cvt_w_kernel<<<(3 * C_DIM * C_DIM + 255) / 256, 256, 0, stream>>>(
      qkvw, projw, wq, wp);

  // 2. GroupNorm -> hnT
  gn_kernel<<<4 * 32, 256, 0, stream>>>(x, gns, gnb, hnT);

  // 3. QKV GEMM -> qT, kT, v
  qkv_gemm_kernel<<<dim3(T_DIM / 128, 3 * C_DIM / 128, 4), 256, 0, stream>>>(
      wq, hnT, qkvb, qT, kT, vv);

  // 4. flash attention -> aT
  attn_kernel<<<4 * NHEAD * (T_DIM / 16 / 8), 256, 0, stream>>>(qT, kT, vv, aT);

  // 5. proj GEMM + residual -> out
  proj_gemm_kernel<<<dim3(T_DIM / 128, C_DIM / 128, 4), 256, 0, stream>>>(
      wp, aT, projb, x, out);
}
